// ShuffleAttention_38053410243413
// MI455X (gfx1250) — matrix operand order repaired and checked by reading, not compiled, not run on hardware
//
#include <hip/hip_runtime.h>
#include <hip/hip_bf16.h>

typedef __attribute__((ext_vector_type(16))) __bf16 v16bf;
typedef __attribute__((ext_vector_type(8)))  __bf16 v8bf;
typedef __attribute__((ext_vector_type(8)))  float  v8f;
typedef __attribute__((ext_vector_type(4)))  unsigned int v4u;
typedef __attribute__((ext_vector_type(8)))  int v8i;
typedef __attribute__((ext_vector_type(4)))  int v4i;

#define DIM    256
#define HEADS  8
#define HD     32          // head dim
#define WS     7
#define WIN    49
#define NPAD   64
#define IMG    112
#define HW     (IMG*IMG)   // 12544
#define REG    56
#define SCALEF 0.17677669529663687f   // 32^-0.5
#define NQKV   (768*256)              // w_qkv elements
#define NPROJ  (256*256)              // w_proj elements
#define MIDELEMS ((size_t)16*DIM*HW)  // bf16 scratch elements

union V16 { v16bf v; v8bf h[2]; };

static __device__ __forceinline__ unsigned short bfb(float f) {
  __bf16 b = (__bf16)f;
  union { __bf16 b; unsigned short s; } u; u.b = b;
  return u.s;
}

static __device__ __forceinline__ v8f v8f_zero() {
  v8f z;
  #pragma unroll
  for (int i = 0; i < 8; ++i) z[i] = 0.0f;
  return z;
}

// =====================================================================
// Kernel 0: pack fp32 weights -> bf16 (row-major [M][K]) in scratch.
// =====================================================================
__global__ void pack_weights_kernel(const float* __restrict__ wq,
                                    const float* __restrict__ wp,
                                    __bf16* __restrict__ wqb,
                                    __bf16* __restrict__ wpb) {
  int idx = blockIdx.x * 256 + threadIdx.x;       // 262144 threads
  if (idx < NQKV) {
    wqb[idx] = (__bf16)wq[idx];
  } else {
    int j = idx - NQKV;                           // < NPROJ
    wpb[j] = (__bf16)wp[j];
  }
}

// =====================================================================
// Kernel 1: fused QKV projection + windowed attention for ONE 7x7 window.
// grid = 4096 windows, 256 threads (8 waves).
// LDS: Xs[64px][256ch] | Qt[64][256] | Kt[64][256] | Vs[256][64] | Ss[8][64][64]f32
// mid is written PIXEL-MAJOR: [b][pixel][ch] so the proj kernel can stage
// its activation tile with a single contiguous TDM copy.
// =====================================================================
__global__ void shuffle_attn_qkv_kernel(const float* __restrict__ x,
                                        const __bf16* __restrict__ wqb,
                                        __bf16* __restrict__ mid) {
  extern __shared__ char smem[];
  __bf16* Xs = (__bf16*)(smem);            // [px][ch]
  __bf16* Qt = (__bf16*)(smem + 32768);    // [px][ch]  (transposed Q)
  __bf16* Kt = (__bf16*)(smem + 65536);    // [px][ch]  (transposed K)
  __bf16* Vs = (__bf16*)(smem + 98304);    // [ch][px]  (row-major V)
  float*  Ss = (float*)(smem + 131072);    // [head][64][64]

  const int tid  = threadIdx.x;
  const int lane = tid & 31;
  const int wave = tid >> 5;
  const int l15  = lane & 15;
  const int hi8  = (lane >> 4) << 3;       // 0 / 8
  const int hi16 = (lane >> 4) << 4;       // 0 / 16

  // ---- decode window position
  int win  = blockIdx.x;
  int b    = win >> 8;
  int rem  = win & 255;
  int regn = rem >> 6;
  int wrem = rem & 63;
  int wy = wrem >> 3, wx = wrem & 7;
  int row0 = (regn >> 1) * REG + wy * WS;
  int col0 = (regn & 1)  * REG + wx * WS;
  const float* xb = x + (size_t)b * DIM * HW;

  // ---- stage window pixels -> Xs[px][ch] bf16 (coalesced reads, LDS scatter)
  for (int e = tid; e < DIM * NPAD; e += 256) {
    int c = e >> 6, p = e & 63;
    __bf16 v = (__bf16)0.0f;
    if (p < WIN) {
      int py = p / WS, px = p - py * WS;
      v = (__bf16)xb[(size_t)c * HW + (row0 + py) * IMG + (col0 + px)];
    }
    Xs[p * DIM + c] = v;
  }
  __syncthreads();

  // ---- QKV GEMM: [768x256] * [256x64] ; each wave owns 6 M-tiles
  for (int j = 0; j < 6; ++j) {
    int m = wave * 6 + j;                   // 0..47
    int rowbase = m * 16;
    int mtx = rowbase >> 8;                 // 0=Q 1=K 2=V
    int mr0 = (rowbase & 255) + hi8;
    v8f acc[4];
    #pragma unroll
    for (int n = 0; n < 4; ++n) acc[n] = v8f_zero();

    int M = m * 16 + l15;
    const __bf16* wrow = wqb + (size_t)M * DIM;
    for (int kb = 0; kb < 8; ++kb) {
      __builtin_prefetch(wrow + kb * 32 + 32, 0, 0);
      V16 a;
      a.h[0] = *(const v8bf*)(wrow + kb * 32 + hi16);       // K 0..7
      a.h[1] = *(const v8bf*)(wrow + kb * 32 + hi16 + 8);   // K 8..15
      #pragma unroll
      for (int n = 0; n < 4; ++n) {
        int col = n * 16 + l15;
        v16bf bf = *(const v16bf*)(Xs + col * DIM + kb * 32 + hi16);
        acc[n] = __builtin_amdgcn_wmma_f32_16x16x32_bf16(
                     false, a.v, false, bf, (short)0, acc[n], false, false);
      }
    }
    // scatter: Q,K transposed [px][ch] (paired b32 stores); V row-major
    if (mtx == 2) {
      #pragma unroll
      for (int n = 0; n < 4; ++n) {
        int col = n * 16 + l15;
        #pragma unroll
        for (int r = 0; r < 8; ++r)
          Vs[(mr0 + r) * NPAD + col] = (__bf16)acc[n][r];
      }
    } else {
      __bf16* T = (mtx == 0) ? Qt : Kt;
      #pragma unroll
      for (int n = 0; n < 4; ++n) {
        int col = n * 16 + l15;
        #pragma unroll
        for (int r = 0; r < 8; r += 2) {
          unsigned pk = (unsigned)bfb(acc[n][r]) |
                        ((unsigned)bfb(acc[n][r + 1]) << 16);
          *(unsigned*)(T + col * DIM + mr0 + r) = pk;       // mr0+r even
        }
      }
    }
  }
  __syncthreads();

  // ---- per-head attention (head == wave)
  const int h = wave;
  const __bf16* Vh = Vs + h * HD * NPAD;
  float* S = Ss + h * NPAD * NPAD;

  // S = scale * Q^T K  (M=64 q, N=64 k, K=32 d)
  for (int mi = 0; mi < 4; ++mi) {
    int Mq = mi * 16 + l15;
    V16 a;
    const __bf16* qrow = Qt + Mq * DIM + h * HD;
    a.h[0] = *(const v8bf*)(qrow + hi16);
    a.h[1] = *(const v8bf*)(qrow + hi16 + 8);
    #pragma unroll
    for (int ni = 0; ni < 4; ++ni) {
      int px = ni * 16 + l15;
      v16bf bf = *(const v16bf*)(Kt + px * DIM + h * HD + hi16);
      v8f acc = v8f_zero();
      acc = __builtin_amdgcn_wmma_f32_16x16x32_bf16(
                false, a.v, false, bf, (short)0, acc, false, false);
      #pragma unroll
      for (int r = 0; r < 8; ++r)
        S[(mi * 16 + r + hi8) * NPAD + px] = acc[r] * SCALEF;
    }
  }
  __syncthreads();

  // softmax over k (49 valid); rewrite rows in place as bf16 (stride 128)
  for (int q = lane; q < NPAD; q += 32) {
    float* row = S + q * NPAD;
    __bf16* arow = (__bf16*)row;
    if (q < WIN) {
      float mx = -1e30f;
      for (int k = 0; k < WIN; ++k) mx = fmaxf(mx, row[k]);
      float sum = 0.0f;
      for (int k = 0; k < WIN; ++k) sum += __expf(row[k] - mx);
      float inv = 1.0f / sum;
      for (int k = 0; k < NPAD; ++k) {            // ascending: safe in-place
        float p = (k < WIN) ? __expf(row[k] - mx) * inv : 0.0f;
        arow[k] = (__bf16)p;
      }
    } else {
      for (int k = 0; k < NPAD; ++k) arow[k] = (__bf16)0.0f;
    }
  }
  __syncthreads();

  // out = attn * V^T  (M=64 q, N=32 d, K=64 k)  -> mid[b][pixel][ch]
  const __bf16* Abf = (const __bf16*)S;           // row stride 128 bf16
  __bf16* midb = mid + (size_t)b * HW * DIM;
  for (int mi = 0; mi < 4; ++mi) {
    int Mq = mi * 16 + l15;
    V16 a0, a1;
    const __bf16* arow = Abf + Mq * 128;
    a0.h[0] = *(const v8bf*)(arow + hi16);
    a0.h[1] = *(const v8bf*)(arow + hi16 + 8);
    a1.h[0] = *(const v8bf*)(arow + 32 + hi16);
    a1.h[1] = *(const v8bf*)(arow + 32 + hi16 + 8);
    #pragma unroll
    for (int ni = 0; ni < 2; ++ni) {
      int d = ni * 16 + l15;
      const __bf16* vrow = Vh + d * NPAD;
      v16bf b0 = *(const v16bf*)(vrow + hi16);
      v16bf b1 = *(const v16bf*)(vrow + 32 + hi16);
      v8f acc = v8f_zero();
      acc = __builtin_amdgcn_wmma_f32_16x16x32_bf16(
                false, a0.v, false, b0, (short)0, acc, false, false);
      acc = __builtin_amdgcn_wmma_f32_16x16x32_bf16(
                false, a1.v, false, b1, (short)0, acc, false, false);
      #pragma unroll
      for (int r = 0; r < 8; ++r) {
        int q = mi * 16 + r + hi8;
        if (q < WIN) {
          int c = h * HD + d;
          int py = q / WS, px = q - py * WS;
          int pix = (row0 + py) * IMG + (col0 + px);
          midb[(size_t)pix * DIM + c] = (__bf16)acc[r];
        }
      }
    }
  }
}

// =====================================================================
// Kernel 2: projection GEMM  out = W_proj[256x256] * mid + bias (fp32 out)
// grid = 6272 blocks of 256 thr; 32-px column slab per block.
// Activation tile [32 px][256 ch] is contiguous in mid -> staged with one
// Tensor Data Mover copy (tensor_load_to_lds + s_wait_tensorcnt).
// =====================================================================
__global__ void shuffle_attn_proj_kernel(const __bf16* __restrict__ mid,
                                         const __bf16* __restrict__ wpb,
                                         const float* __restrict__ b_proj,
                                         float* __restrict__ out) {
  __shared__ __bf16 Xs[32 * DIM];                 // [px][ic], 16 KB
  const int tid = threadIdx.x, lane = tid & 31, wave = tid >> 5;
  const int l15 = lane & 15;
  const int hi8 = (lane >> 4) << 3;
  const int hi16 = (lane >> 4) << 4;
  int pxBase = blockIdx.x * 32;
  int b   = pxBase / HW;                          // blocks never straddle batches
  int pos = pxBase - b * HW;

  // ---- TDM: one contiguous 16 KB tile  mid[b][pos..pos+31][0..255] -> Xs
  if (wave == 0) {
    unsigned long long ga =
        (unsigned long long)(mid + ((size_t)b * HW + pos) * DIM);
    unsigned lds_off = (unsigned)(size_t)(&Xs[0]);
    v4u g0;
    g0[0] = 1u;                                   // count=1, no gather
    g0[1] = lds_off;                              // D#.lds_addr (bytes)
    g0[2] = (unsigned)ga;                         // global_addr[31:0]
    g0[3] = ((unsigned)(ga >> 32) & 0x01FFFFFFu)  // global_addr[56:32]
            | (2u << 30);                         // type=2 (image)
    v8i g1;
    g1[0] = 0x00010000;                           // data_size=1 (2 bytes)
    g1[1] = (int)(8192u << 16);                   // tensor_dim0[15:0]=8192
    g1[2] = (int)(1u << 16);                      // tensor_dim1=1
    g1[3] = (int)(8192u << 16);                   // tile_dim0=8192
    g1[4] = 1;                                    // tile_dim1=1
    g1[5] = 8192;                                 // tensor_dim0_stride
    g1[6] = 0;
    g1[7] = 0;
    v4i gz4;
    gz4[0] = 0; gz4[1] = 0; gz4[2] = 0; gz4[3] = 0;
    v8i gz8;
    #pragma unroll
    for (int i = 0; i < 8; ++i) gz8[i] = 0;
    __builtin_amdgcn_tensor_load_to_lds(g0, g1, gz4, gz4, gz8, 0);
    __builtin_amdgcn_s_wait_tensorcnt(0);
  }
  __syncthreads();

  v8f acc[2][2];
  #pragma unroll
  for (int i = 0; i < 2; ++i) {
    #pragma unroll
    for (int j = 0; j < 2; ++j) acc[i][j] = v8f_zero();
  }

  for (int kb = 0; kb < 8; ++kb) {
    v16bf bf0 = *(const v16bf*)(Xs + (l15)      * DIM + kb * 32 + hi16);
    v16bf bf1 = *(const v16bf*)(Xs + (16 + l15) * DIM + kb * 32 + hi16);
    #pragma unroll
    for (int mj = 0; mj < 2; ++mj) {
      int M = (wave * 2 + mj) * 16 + l15;
      const __bf16* wrow = wpb + (size_t)M * DIM + kb * 32;
      __builtin_prefetch(wrow + 32, 0, 0);
      V16 a;
      a.h[0] = *(const v8bf*)(wrow + hi16);
      a.h[1] = *(const v8bf*)(wrow + hi16 + 8);
      acc[mj][0] = __builtin_amdgcn_wmma_f32_16x16x32_bf16(
                       false, a.v, false, bf0, (short)0, acc[mj][0], false, false);
      acc[mj][1] = __builtin_amdgcn_wmma_f32_16x16x32_bf16(
                       false, a.v, false, bf1, (short)0, acc[mj][1], false, false);
    }
  }

  #pragma unroll
  for (int mj = 0; mj < 2; ++mj) {
    #pragma unroll
    for (int ni = 0; ni < 2; ++ni) {
      #pragma unroll
      for (int r = 0; r < 8; ++r) {
        int oc = (wave * 2 + mj) * 16 + r + hi8;
        int p  = ni * 16 + l15;
        out[((size_t)(b * DIM + oc)) * HW + pos + p] = acc[mj][ni][r] + b_proj[oc];
      }
    }
  }
}

// =====================================================================
extern "C" void kernel_launch(void* const* d_in, const int* in_sizes, int n_in,
                              void* d_out, int out_size, void* d_ws, size_t ws_size,
                              hipStream_t stream) {
  const float* x      = (const float*)d_in[0];
  const float* w_qkv  = (const float*)d_in[1];
  const float* w_proj = (const float*)d_in[2];
  const float* b_proj = (const float*)d_in[3];
  float* out = (float*)d_out;

  __bf16* mid = (__bf16*)d_ws;                          // 98 MB bf16, [b][pix][ch]
  __bf16* wqb = (__bf16*)((char*)d_ws + MIDELEMS * 2);  // 768x256 bf16
  __bf16* wpb = wqb + NQKV;                             // 256x256 bf16

  pack_weights_kernel<<<(NQKV + NPROJ) / 256, 256, 0, stream>>>(w_qkv, w_proj,
                                                                wqb, wpb);

  const size_t lds1 = 262144;                           // 256 KB dynamic LDS
  (void)hipFuncSetAttribute((const void*)shuffle_attn_qkv_kernel,
                            hipFuncAttributeMaxDynamicSharedMemorySize, (int)lds1);
  shuffle_attn_qkv_kernel<<<16 * 4 * 64, 256, lds1, stream>>>(x, wqb, mid);

  shuffle_attn_proj_kernel<<<16 * HW / 32, 256, 0, stream>>>(mid, wpb, b_proj, out);
}
